// MyMonotoneNN_strict_dim2_36953898615397
// MI455X (gfx1250) — matrix operand order repaired and checked by reading, not compiled, not run on hardware
//
#include <hip/hip_runtime.h>

typedef _Float16 h16;
typedef __attribute__((ext_vector_type(2)))  _Float16 h2;
typedef __attribute__((ext_vector_type(16))) _Float16 v16h;
typedef __attribute__((ext_vector_type(8)))  float    v8f;

union V16H { v16h v; h2 p[8]; };

#define WROWS 4

// One wave32 handles WROWS batch rows, both dims, all 128 quadrature points.
// Layer2 (64x64) runs on v_wmma_f32_16x16x32_f16; A operand is generated
// in-register in the ISA-documented A layout, B operand (w2^T) likewise.
// Layer3 dot is reduced with a butterfly reduce-scatter so ELU runs once per
// lane instead of 8x redundantly.
__global__ __launch_bounds__(256) void monotone_nn_wmma_kernel(
    const float* __restrict__ z,   const float* __restrict__ u,
    const float* __restrict__ w01, const float* __restrict__ b01,
    const float* __restrict__ w02, const float* __restrict__ b02,
    const float* __restrict__ w03, const float* __restrict__ b03,
    const float* __restrict__ w11, const float* __restrict__ b11,
    const float* __restrict__ w12, const float* __restrict__ b12,
    const float* __restrict__ w13, const float* __restrict__ b13,
    const float* __restrict__ bias, float* __restrict__ out, int Bn)
{
    const int lane = threadIdx.x & 31;
    const int g    = lane >> 4;      // half-wave group (0: lanes 0-15, 1: 16-31)
    const int m16  = lane & 15;      // M index (samples) for A, N index (k_out) for B/C/D
    const bool bit0 = (lane & 1) != 0;
    const bool bit1 = (lane & 2) != 0;
    const bool bit2 = (lane & 4) != 0;
    const int wave = blockIdx.x * (blockDim.x >> 5) + (threadIdx.x >> 5);
    const int rowbase = wave * WROWS;
    if (rowbase >= Bn) return;

    float rowAcc[WROWS];
#pragma unroll
    for (int r = 0; r < WROWS; ++r) rowAcc[r] = 0.0f;

    for (int d = 0; d < 2; ++d) {
        const float* w1 = d ? w11 : w01;
        const float* b1 = d ? b11 : b01;
        const float* w2 = d ? w12 : w02;
        const float* b2 = d ? b12 : b02;
        const float* w3 = d ? w13 : w03;
        const float  b3 = d ? b13[0] : b03[0];

        // ---- Layer-1 params, packed per lane to match the f16 A layout ----
        // A (16x32, MxK): lane half g covers K pairs {0..7,16..23}+8g per vgpr table.
        h2 w1p[2][8], b1p[2][8];
#pragma unroll
        for (int a = 0; a < 2; ++a)
#pragma unroll
            for (int j = 0; j < 8; ++j) {
                int k = ((j >> 2) << 4) + ((j & 3) << 1) + (g << 3) + 32 * a;
                h2 wv = { (h16)w1[k], (h16)w1[k + 1] };
                h2 bv = { (h16)b1[k], (h16)b1[k + 1] };
                w1p[a][j] = wv;
                b1p[a][j] = bv;
            }

        // ---- B operands: w2^T in WMMA B layout (32x16, K pairs {0..7,16..23}+8g per half) ----
        // 4 column blocks (k_out) x 2 K-chunks (h_in 0-31, 32-63).
        V16H Bop[4][2];
        float b2l[4], w3l[4];
#pragma unroll
        for (int nb = 0; nb < 4; ++nb) {
            int kout = nb * 16 + m16;
            b2l[nb] = b2[kout];
            w3l[nb] = w3[kout];
#pragma unroll
            for (int c = 0; c < 2; ++c)
#pragma unroll
                for (int j = 0; j < 8; ++j) {
                    int h = 32 * c + 8 * g + ((j >> 2) << 4) + ((j & 3) << 1);
                    h2 bv = { (h16)w2[kout * 64 + h], (h16)w2[kout * 64 + h + 1] };
                    Bop[nb][c].p[j] = bv;
                }
        }

        h2 slope = { (h16)0.01f, (h16)0.01f };

#pragma unroll
        for (int rr = 0; rr < WROWS; ++rr) {
            int brow = rowbase + rr;
            float zc = fmaxf(z[brow * 2 + d], -5.0f);
            float ds = (zc + 5.0f) * (1.0f / 127.0f);
            const float* urow = u + (brow * 2 + d) * 127;

            float osum = 0.0f;  // per-lane: sum of elu(sample r = lane&7 (+8g)) over tiles
            for (int t = 0; t < 8; ++t) {
                int n = t * 16 + m16;                 // quadrature point index
                float jit = 0.0f;
                if (n < 127) jit = urow[n];
                float x = fmaf(ds, (float)n + jit, -5.0f);
                h16 xh = (h16)x;
                h2 xh2 = { xh, xh };

                // Layer 1 in packed f16, directly into A layout
                V16H A[2];
#pragma unroll
                for (int a = 0; a < 2; ++a)
#pragma unroll
                    for (int j = 0; j < 8; ++j) {
                        h2 v = xh2 * w1p[a][j] + b1p[a][j];
                        v = __builtin_elementwise_max(v, v * slope);  // leaky relu
                        A[a].p[j] = v;
                    }

                // Layer 2 via WMMA (bias folded into C init) + layer-3 partial dot
                float partial[8];
#pragma unroll
                for (int r = 0; r < 8; ++r) partial[r] = 0.0f;

#pragma unroll
                for (int nb = 0; nb < 4; ++nb) {
                    v8f c;
#pragma unroll
                    for (int r = 0; r < 8; ++r) c[r] = b2l[nb];
                    c = __builtin_amdgcn_wmma_f32_16x16x32_f16(
                            false, A[0].v, false, Bop[nb][0].v, (short)0, c, false, false);
                    c = __builtin_amdgcn_wmma_f32_16x16x32_f16(
                            false, A[1].v, false, Bop[nb][1].v, (short)0, c, false, false);
#pragma unroll
                    for (int r = 0; r < 8; ++r) {
                        float hv = c[r];
                        hv = fmaxf(hv, hv * 0.01f);                  // leaky relu
                        partial[r] = fmaf(hv, w3l[nb], partial[r]);  // dot with w3
                    }
                }

                // Butterfly reduce-scatter across the 16 lanes of each half-wave:
                // map r-bit k onto lane-bit k. After 3 stages + mask-8 add, lane l
                // holds S[r = l&7] = full 16-lane sum (lanes 8-15 duplicate 0-7).
                float w4[4];
#pragma unroll
                for (int i = 0; i < 4; ++i) {
                    float keep = bit0 ? partial[2 * i + 1] : partial[2 * i];
                    float send = bit0 ? partial[2 * i]     : partial[2 * i + 1];
                    w4[i] = keep + __shfl_xor(send, 1, 32);
                }
                float x2[2];
#pragma unroll
                for (int j = 0; j < 2; ++j) {
                    float keep = bit1 ? w4[2 * j + 1] : w4[2 * j];
                    float send = bit1 ? w4[2 * j]     : w4[2 * j + 1];
                    x2[j] = keep + __shfl_xor(send, 2, 32);
                }
                {
                    float keep = bit2 ? x2[1] : x2[0];
                    float send = bit2 ? x2[0] : x2[1];
                    float y = keep + __shfl_xor(send, 4, 32);
                    y += __shfl_xor(y, 8, 32);        // full 16-lane sum
                    // ELU once per lane (sample m = (lane&7) + 8g, duplicated x2)
                    float tt = y + b3;
                    float o = tt > 0.0f ? tt : (__expf(tt) - 1.0f);
                    osum += o;
                }
            }
            // Sum osum over all 32 lanes: every sample counted exactly twice.
            float p = osum;
            p += __shfl_xor(p, 1, 32);
            p += __shfl_xor(p, 2, 32);
            p += __shfl_xor(p, 4, 32);
            p += __shfl_xor(p, 8, 32);
            p += __shfl_xor(p, 16, 32);
            // sum_n (o_n + 1) * ds = ds * (sum_n o_n + 128); p = 2 * sum_n o_n
            rowAcc[rr] += ds * fmaf(0.5f, p, 128.0f);
        }
    }

    if (lane == 0) {
        float bs = bias[0];
#pragma unroll
        for (int r = 0; r < WROWS; ++r)
            if (rowbase + r < Bn) out[rowbase + r] = rowAcc[r] + bs;
    }
}

extern "C" void kernel_launch(void* const* d_in, const int* in_sizes, int n_in,
                              void* d_out, int out_size, void* d_ws, size_t ws_size,
                              hipStream_t stream) {
    (void)n_in; (void)d_ws; (void)ws_size; (void)out_size;
    const float* z    = (const float*)d_in[0];
    const float* u    = (const float*)d_in[1];
    const float* w01  = (const float*)d_in[2];
    const float* b01  = (const float*)d_in[3];
    const float* w02  = (const float*)d_in[4];
    const float* b02  = (const float*)d_in[5];
    const float* w03  = (const float*)d_in[6];
    const float* b03  = (const float*)d_in[7];
    const float* w11  = (const float*)d_in[8];
    const float* b11  = (const float*)d_in[9];
    const float* w12  = (const float*)d_in[10];
    const float* b12  = (const float*)d_in[11];
    const float* w13  = (const float*)d_in[12];
    const float* b13  = (const float*)d_in[13];
    const float* bias = (const float*)d_in[14];
    float* out = (float*)d_out;

    const int Bn = in_sizes[0] / 2;               // 32768 batch rows
    const int waves  = (Bn + WROWS - 1) / WROWS;  // 8192 waves
    const int blocks = (waves + 7) / 8;           // 8 waves / 256-thread block

    monotone_nn_wmma_kernel<<<blocks, 256, 0, stream>>>(
        z, u, w01, b01, w02, b02, w03, b03,
        w11, b11, w12, b12, w13, b13, bias, out, Bn);
}
